// SwinTransformerBlock3D_5171140625034
// MI455X (gfx1250) — matrix operand order repaired and checked by reading, not compiled, not run on hardware
//
#include <hip/hip_runtime.h>
#include <math.h>

// ---------------------------------------------------------------------------
// Video Swin 3D block, MI455X (gfx1250, wave32, WMMA bf16 16x16x32).
// GEMMs on v_wmma_f32_16x16x32_bf16; weight/activation panels staged in LDS
// via CDNA5 async global->LDS (ASYNCcnt); LN/softmax/GELU in fp32 VALU.
// ---------------------------------------------------------------------------

typedef __bf16 bf16;
typedef bf16  v16bf __attribute__((ext_vector_type(16)));
typedef float v8f   __attribute__((ext_vector_type(8)));

__device__ __forceinline__ bf16 f2bf(float f) {
  unsigned u = __builtin_bit_cast(unsigned, f);
  unsigned r = (u + 0x7FFFu + ((u >> 16) & 1u)) >> 16;
  unsigned short s = (unsigned short)r;
  return __builtin_bit_cast(bf16, s);
}
__device__ __forceinline__ v8f zero8() {
  v8f z = {0.f, 0.f, 0.f, 0.f, 0.f, 0.f, 0.f, 0.f};
  return z;
}

// ---- CDNA5 async global->LDS copy (16B per lane), tracked by ASYNCcnt ------
__device__ __forceinline__ void async_g2l_b128(unsigned lds_off, const void* gptr) {
  asm volatile("global_load_async_to_lds_b128 %0, %1, off"
               :: "v"(lds_off), "v"(gptr) : "memory");
}
__device__ __forceinline__ void s_wait_async0() {
  asm volatile("s_wait_asynccnt 0x0" ::: "memory");
}
__device__ __forceinline__ unsigned lds_addr_of(const void* p) {
  return (unsigned)(size_t)p;  // addrspace(3) pointer -> LDS byte offset
}

// A-fragment (16x32 bf16) from row-major [M][lda].
// lane m = lane&15; lanes 0-15 hold K 0-7 & 16-23, lanes 16-31 hold K 8-15 & 24-31.
__device__ __forceinline__ v16bf load_a_frag(const bf16* base, int lda) {
  int lane = threadIdx.x & 31;
  const bf16* p = base + (lane & 15) * lda;
  int kb = (lane >> 4) << 3;  // 0 or 8
  v16bf a;
#pragma unroll
  for (int j = 0; j < 4; ++j) {
    int k = kb + 2 * j;
    a[2 * j]     = p[k];
    a[2 * j + 1] = p[k + 1];
    a[8 + 2 * j]     = p[16 + k];
    a[8 + 2 * j + 1] = p[16 + k + 1];
  }
  return a;
}

// B-fragment (32x16 bf16) from B^T stored row-major [N][ldb] (element(k,n)=bt[n*ldb+k]).
// lane n = lane&15; lanes 0-15 hold K 0-7 & 16-23, lanes 16-31 hold K 8-15 & 24-31.
__device__ __forceinline__ v16bf load_bt_frag(const bf16* bt, int ldb) {
  int lane = threadIdx.x & 31;
  const bf16* p = bt + (lane & 15) * ldb + ((lane >> 4) << 3);
  v16bf b;
#pragma unroll
  for (int j = 0; j < 16; ++j) b[j] = p[j < 8 ? j : 8 + j];
  return b;
}

__device__ __forceinline__ v8f wmma_bf16(v16bf a, v16bf b, v8f c) {
  return __builtin_amdgcn_wmma_f32_16x16x32_bf16(false, a, false, b, (short)0, c,
                                                 false, false);
}

__device__ __forceinline__ float wave_sum32(float v) {
  v += __shfl_xor(v, 1);  v += __shfl_xor(v, 2);
  v += __shfl_xor(v, 4);  v += __shfl_xor(v, 8);
  v += __shfl_xor(v, 16);
  return v;
}
__device__ __forceinline__ float half_sum16(float v) {
  v += __shfl_xor(v, 1); v += __shfl_xor(v, 2);
  v += __shfl_xor(v, 4); v += __shfl_xor(v, 8);
  return v;
}
__device__ __forceinline__ float half_max16(float v) {
  v = fmaxf(v, __shfl_xor(v, 1)); v = fmaxf(v, __shfl_xor(v, 2));
  v = fmaxf(v, __shfl_xor(v, 4)); v = fmaxf(v, __shfl_xor(v, 8));
  return v;
}

// window-layout row -> flat token index (window reverse + un-roll(+1,+3,+3))
__device__ __forceinline__ size_t token_of_row(int rowg) {
  int win = rowg / 98, n = rowg % 98;
  int b  = win >> 8, rem = win & 255;
  int wd = rem >> 6, wh = (rem >> 3) & 7, ww = rem & 7;
  int dz = n / 49, hz = (n % 49) / 7, wz = n % 7;
  int d = (wd * 2 + dz + 1) & 7;
  int h = (wh * 7 + hz + 3) % 56;
  int w = (ww * 7 + wz + 3) % 56;
  return ((size_t)((b * 8 + d) * 56 + h)) * 56 + w;
}

// ---------------------------------------------------------------------------
// K0: convert weights to bf16 (fold q-scale 1/sqrt(32) into Wq rows / bq)
// ---------------------------------------------------------------------------
__global__ void k_prep(const float* qkv_w, const float* qkv_b, const float* proj_w,
                       const float* fc1_w, const float* fc2_w,
                       bf16* wqkv, float* bqkv, bf16* wproj, bf16* w1, bf16* w2) {
  int i = blockIdx.x * blockDim.x + threadIdx.x;
  const float SC = 0.17677669529663687f;  // 32^-0.5
  if (i < 49152) {
    int o = i >> 7;
    wqkv[i] = f2bf(qkv_w[i] * (o < 128 ? SC : 1.f));
  } else if (i < 65536) {
    int j = i - 49152; wproj[j] = f2bf(proj_w[j]);
  } else if (i < 131072) {
    int j = i - 65536; w1[j] = f2bf(fc1_w[j]);
  } else if (i < 196608) {
    int j = i - 131072; w2[j] = f2bf(fc2_w[j]);
  } else if (i < 196992) {
    int j = i - 196608; bqkv[j] = qkv_b[j] * (j < 128 ? SC : 1.f);
  }
}

// ---------------------------------------------------------------------------
// K1: relative-position bias table  Bias[head][98][98]
// ---------------------------------------------------------------------------
__global__ void k_bias(const float* rel_bias, float* btab) {
  int t = blockIdx.x * blockDim.x + threadIdx.x;
  if (t >= 98 * 98) return;
  int i = t / 98, j = t % 98;
  int di = i / 49, hi = (i % 49) / 7, wi = i % 7;
  int dj = j / 49, hj = (j % 49) / 7, wj = j % 7;
  int idx = (di - dj + 1) * 169 + (hi - hj + 6) * 13 + (wi - wj + 6);
#pragma unroll
  for (int h = 0; h < 4; ++h) btab[(h * 98 + i) * 98 + j] = rel_bias[idx * 4 + h];
}

// ---------------------------------------------------------------------------
// K2: LN1 + cyclic shift + window partition -> Xw bf16 (window layout).
// One wave32 per token; 8 waves / block.
// ---------------------------------------------------------------------------
__global__ void __launch_bounds__(256) k_ln1(const float* x, const float* g,
                                             const float* b, bf16* xw) {
  int wv = threadIdx.x >> 5, lane = threadIdx.x & 31;
  int token = blockIdx.x * 8 + wv;  // 25088*8 = 200704 exactly
  const float* row = x + (size_t)token * 128;
  float4 v = ((const float4*)row)[lane];
  float mean = wave_sum32(v.x + v.y + v.z + v.w) * (1.f / 128.f);
  float dx = v.x - mean, dy = v.y - mean, dz = v.z - mean, dw = v.w - mean;
  float var = wave_sum32(dx * dx + dy * dy + dz * dz + dw * dw) * (1.f / 128.f);
  float inv = rsqrtf(var + 1e-5f);
  int w = token % 56, h = (token / 56) % 56, d = (token / 3136) & 7, bb = token / 25088;
  int ds = (d + 7) & 7, hs = (h + 53) % 56, wsx = (w + 53) % 56;
  int win = ((bb * 4 + (ds >> 1)) * 8 + hs / 7) * 8 + wsx / 7;
  int n = (ds & 1) * 49 + (hs % 7) * 7 + (wsx % 7);
  bf16* dst = xw + ((size_t)win * 98 + n) * 128 + lane * 4;
  float4 gg = ((const float4*)g)[lane];
  float4 bv = ((const float4*)b)[lane];
  dst[0] = f2bf(dx * inv * gg.x + bv.x);
  dst[1] = f2bf(dy * inv * gg.y + bv.y);
  dst[2] = f2bf(dz * inv * gg.z + bv.z);
  dst[3] = f2bf(dw * inv * gg.w + bv.w);
}

// ---------------------------------------------------------------------------
// K3: QKV GEMM  (200704x128) @ (384x128)^T + bias -> QKV bf16 [row][384]
// grid (3136, 6); block 128 = 4 waves; weight panel async-staged in LDS.
// Inner loop: batch all 4 B ds_loads before the 4 WMMAs (single dscnt wait).
// ---------------------------------------------------------------------------
__global__ void __launch_bounds__(128) k_qkv(const bf16* xw, const bf16* wqkv,
                                             const float* bqkv, bf16* qkv) {
  __shared__ __attribute__((aligned(16))) bf16 Wp[64 * 128];  // 16KB B panel
  int tid = threadIdx.x, wv = tid >> 5, lane = tid & 31;
  int m0 = blockIdx.x * 64 + wv * 16;
  int n0 = blockIdx.y * 64;

  // async-stage weight panel rows [n0, n0+64) x K128 into LDS (1024 x 16B)
  {
    const char* src = (const char*)(wqkv + (size_t)n0 * 128);
    unsigned dst = lds_addr_of(Wp);
#pragma unroll
    for (int i = 0; i < 8; ++i) {
      unsigned off = (unsigned)(tid + i * 128) * 16u;
      async_g2l_b128(dst + off, src + off);
    }
  }
  // prefetch this block's A rows while the async DMA runs
  if (tid < 64) __builtin_prefetch(xw + (size_t)(blockIdx.x * 64 + tid) * 128, 0, 0);
  s_wait_async0();
  __syncthreads();

  v8f acc[4] = {zero8(), zero8(), zero8(), zero8()};
  const bf16* arow = xw + (size_t)m0 * 128;
#pragma unroll
  for (int k0 = 0; k0 < 128; k0 += 32) {
    v16bf a = load_a_frag(arow + k0, 128);
    v16bf bbr[4];
#pragma unroll
    for (int nt = 0; nt < 4; ++nt)
      bbr[nt] = load_bt_frag(Wp + (nt * 16) * 128 + k0, 128);
#pragma unroll
    for (int nt = 0; nt < 4; ++nt)
      acc[nt] = wmma_bf16(a, bbr[nt], acc[nt]);
  }
  int hi = lane >> 4, col0 = lane & 15;
#pragma unroll
  for (int nt = 0; nt < 4; ++nt) {
    int colg = n0 + nt * 16 + col0;
    float bias = bqkv[colg];
#pragma unroll
    for (int r = 0; r < 8; ++r) {
      int rowg = m0 + r + hi * 8;
      qkv[(size_t)rowg * 384 + colg] = f2bf(acc[nt][r] + bias);
    }
  }
}

// ---------------------------------------------------------------------------
// K4: attention per (window, head). grid (2048,4), block 224 = 7 waves.
// Q/K rows async-copied to LDS; S = Q K^T, bias+mask+softmax, O = P V.
// ---------------------------------------------------------------------------
__global__ void __launch_bounds__(224) k_attn(const bf16* qkv, const float* btab,
                                              const float* maskm, bf16* attout) {
  __shared__ __attribute__((aligned(16))) bf16 Qs[112 * 32];
  __shared__ __attribute__((aligned(16))) bf16 Ks[112 * 32];
  __shared__ __attribute__((aligned(16))) bf16 Vt[32 * 128];   // Vt[c][k], k>=98 zero
  __shared__ __attribute__((aligned(16))) bf16 Ps[112 * 128];  // cols 112..127 zero
  int win = blockIdx.x, head = blockIdx.y;
  int tid = threadIdx.x;
  const bf16* base = qkv + (size_t)win * 98 * 384 + head * 32;
  bf16 zb = f2bf(0.f);

  // zero-pad Q/K rows 98..111
  for (int idx = 98 * 32 + tid; idx < 112 * 32; idx += 224) {
    Qs[idx] = zb; Ks[idx] = zb;
  }
  // async copy valid Q/K rows: 98 rows x 64B = 4 x b128 each
  {
    unsigned qd = lds_addr_of(Qs), kd = lds_addr_of(Ks);
    for (int t = tid; t < 98 * 4; t += 224) {
      int n = t >> 2, seg = t & 3;
      unsigned off = (unsigned)(n * 64 + seg * 16);
      const char* g = (const char*)(base + (size_t)n * 384) + seg * 16;
      async_g2l_b128(qd + off, g);
      async_g2l_b128(kd + off, g + 256);  // +128 bf16 = K block
    }
  }
  // V transposed (regular path) + P zero fill
  for (int idx = tid; idx < 32 * 128; idx += 224) {
    int c = idx >> 7, n = idx & 127;
    Vt[idx] = (n < 98) ? base[(size_t)n * 384 + 256 + c] : zb;
  }
  for (int idx = tid; idx < 112 * 128; idx += 224) Ps[idx] = zb;
  s_wait_async0();
  __syncthreads();

  int wv = tid >> 5, lane = tid & 31;
  int hi = lane >> 4, col0 = lane & 15;

  // S = Q K^T : one 16x112 strip per wave, K-depth 32 -> 1 WMMA per tile
  v16bf aq = load_a_frag(Qs + wv * 16 * 32, 32);
  v8f s[7];
#pragma unroll
  for (int nt = 0; nt < 7; ++nt) {
    v16bf bk = load_bt_frag(Ks + nt * 16 * 32, 32);
    s[nt] = wmma_bf16(aq, bk, zero8());
  }

  // fused bias + mask + softmax in C-layout registers
  const float* bptr = btab + (size_t)head * 98 * 98;
  const float* mptr = maskm + (size_t)(win & 255) * 98 * 98;
#pragma unroll
  for (int r = 0; r < 8; ++r) {
    int row = wv * 16 + r + hi * 8;
    bool rok = row < 98;
    float rmax = -1e30f;
#pragma unroll
    for (int nt = 0; nt < 7; ++nt) {
      int col = nt * 16 + col0;
      float v = s[nt][r];
      if (rok && col < 98)
        v += bptr[row * 98 + col] + mptr[row * 98 + col];
      else
        v = -1e30f;
      s[nt][r] = v;
      rmax = fmaxf(rmax, v);
    }
    rmax = half_max16(rmax);
    float sum = 0.f;
#pragma unroll
    for (int nt = 0; nt < 7; ++nt) {
      float p = __expf(s[nt][r] - rmax);
      s[nt][r] = p;
      sum += p;
    }
    sum = half_sum16(sum);
    float inv = 1.f / fmaxf(sum, 1e-30f);
#pragma unroll
    for (int nt = 0; nt < 7; ++nt)
      Ps[(wv * 16 + r + hi * 8) * 128 + nt * 16 + col0] = f2bf(s[nt][r] * inv);
  }
  __syncthreads();

  // O = P V : 16x32 per wave, K=128 (padded)
  v8f o0 = zero8(), o1 = zero8();
#pragma unroll
  for (int k0 = 0; k0 < 128; k0 += 32) {
    v16bf ap = load_a_frag(Ps + wv * 16 * 128 + k0, 128);
    v16bf b0 = load_bt_frag(Vt + 0 * 16 * 128 + k0, 128);
    v16bf b1 = load_bt_frag(Vt + 1 * 16 * 128 + k0, 128);
    o0 = wmma_bf16(ap, b0, o0);
    o1 = wmma_bf16(ap, b1, o1);
  }
  bf16* outp = attout + (size_t)win * 98 * 128 + head * 32;
#pragma unroll
  for (int r = 0; r < 8; ++r) {
    int row = wv * 16 + r + hi * 8;
    if (row < 98) {
      outp[(size_t)row * 128 + col0]      = f2bf(o0[r]);
      outp[(size_t)row * 128 + 16 + col0] = f2bf(o1[r]);
    }
  }
}

// ---------------------------------------------------------------------------
// K5: proj GEMM + window-reverse scatter + residual -> X1 fp32, then LN2 -> H2 bf16.
// grid 3136, block 256 = 8 waves (4M x 2N of 16x64); wproj async-staged.
// ---------------------------------------------------------------------------
__global__ void __launch_bounds__(256) k_proj(const bf16* attout, const bf16* wproj,
                                              const float* proj_b, const float* x,
                                              const float* g2, const float* b2,
                                              float* x1, bf16* h2) {
  __shared__ __attribute__((aligned(16))) bf16 Wp[128 * 128];  // 32KB
  __shared__ float Xs[64 * 128];                               // 32KB
  int tid = threadIdx.x, wv = tid >> 5, lane = tid & 31;
  int mw = wv >> 1, nw = wv & 1;
  int m0 = blockIdx.x * 64 + mw * 16;
  int n0 = nw * 64;

  // async-stage full proj weight (2048 x 16B over 256 threads)
  {
    const char* src = (const char*)wproj;
    unsigned dst = lds_addr_of(Wp);
#pragma unroll
    for (int i = 0; i < 8; ++i) {
      unsigned off = (unsigned)(tid + i * 256) * 16u;
      async_g2l_b128(dst + off, src + off);
    }
  }
  if (tid < 64) __builtin_prefetch(attout + (size_t)(blockIdx.x * 64 + tid) * 128, 0, 0);
  s_wait_async0();
  __syncthreads();

  v8f acc[4] = {zero8(), zero8(), zero8(), zero8()};
  const bf16* arow = attout + (size_t)m0 * 128;
#pragma unroll
  for (int k0 = 0; k0 < 128; k0 += 32) {
    v16bf a = load_a_frag(arow + k0, 128);
    v16bf bbr[4];
#pragma unroll
    for (int nt = 0; nt < 4; ++nt)
      bbr[nt] = load_bt_frag(Wp + (n0 + nt * 16) * 128 + k0, 128);
#pragma unroll
    for (int nt = 0; nt < 4; ++nt)
      acc[nt] = wmma_bf16(a, bbr[nt], acc[nt]);
  }
  int hi = lane >> 4, col0 = lane & 15;
#pragma unroll
  for (int nt = 0; nt < 4; ++nt) {
    int colg = n0 + nt * 16 + col0;
    float pb = proj_b[colg];
#pragma unroll
    for (int r = 0; r < 8; ++r) {
      int rowl = mw * 16 + r + hi * 8;
      size_t tok = token_of_row(blockIdx.x * 64 + rowl);
      float v = acc[nt][r] + pb + x[tok * 128 + colg];
      x1[tok * 128 + colg] = v;
      Xs[rowl * 128 + colg] = v;
    }
  }
  __syncthreads();
  // LN2 over completed rows: wave wv handles local rows 8*wv .. 8*wv+7
#pragma unroll
  for (int rr = 0; rr < 8; ++rr) {
    int rowl = wv * 8 + rr;
    float4 v = ((const float4*)(Xs + rowl * 128))[lane];
    float mean = wave_sum32(v.x + v.y + v.z + v.w) * (1.f / 128.f);
    float dx = v.x - mean, dy = v.y - mean, dz = v.z - mean, dw = v.w - mean;
    float var = wave_sum32(dx * dx + dy * dy + dz * dz + dw * dw) * (1.f / 128.f);
    float inv = rsqrtf(var + 1e-5f);
    size_t tok = token_of_row(blockIdx.x * 64 + rowl);
    float4 gg = ((const float4*)g2)[lane];
    float4 bv = ((const float4*)b2)[lane];
    bf16* dst = h2 + tok * 128 + lane * 4;
    dst[0] = f2bf(dx * inv * gg.x + bv.x);
    dst[1] = f2bf(dy * inv * gg.y + bv.y);
    dst[2] = f2bf(dz * inv * gg.z + bv.z);
    dst[3] = f2bf(dw * inv * gg.w + bv.w);
  }
}

// ---------------------------------------------------------------------------
// K6: fused MLP: FC1(128->512) + exact GELU (hidden in LDS) + FC2 + residual.
// grid 3136, block 256 = 8 waves; A-tile async-staged and reused 4x.
// ---------------------------------------------------------------------------
__global__ void __launch_bounds__(256) k_mlp(const bf16* h2, const bf16* w1,
                                             const float* b1, const bf16* w2,
                                             const float* b2f, const float* x1,
                                             float* out) {
  __shared__ __attribute__((aligned(16))) bf16 Hs[64 * 128];  // 16KB A tile
  __shared__ __attribute__((aligned(16))) bf16 Ys[64 * 512];  // 64KB hidden
  int tid = threadIdx.x, wv = tid >> 5, lane = tid & 31;
  int mw = wv >> 1, nw = wv & 1;
  int m0 = blockIdx.x * 64 + mw * 16;
  int hi = lane >> 4, col0 = lane & 15;

  // async-stage the 64-token LN2 activation tile (1024 x 16B over 256 threads)
  {
    const char* src = (const char*)(h2 + (size_t)(blockIdx.x * 64) * 128);
    unsigned dst = lds_addr_of(Hs);
#pragma unroll
    for (int i = 0; i < 4; ++i) {
      unsigned off = (unsigned)(tid + i * 256) * 16u;
      async_g2l_b128(dst + off, src + off);
    }
  }
  s_wait_async0();
  __syncthreads();

  const bf16* arow = Hs + (size_t)(mw * 16) * 128;
#pragma unroll
  for (int chunk = 0; chunk < 4; ++chunk) {
    int nbase = chunk * 128 + nw * 64;
    v8f acc[4] = {zero8(), zero8(), zero8(), zero8()};
#pragma unroll
    for (int k0 = 0; k0 < 128; k0 += 32) {
      v16bf a = load_a_frag(arow + k0, 128);
      v16bf bbr[4];
#pragma unroll
      for (int nt = 0; nt < 4; ++nt)
        bbr[nt] = load_bt_frag(w1 + (size_t)(nbase + nt * 16) * 128 + k0, 128);
#pragma unroll
      for (int nt = 0; nt < 4; ++nt)
        acc[nt] = wmma_bf16(a, bbr[nt], acc[nt]);
    }
#pragma unroll
    for (int nt = 0; nt < 4; ++nt) {
      int colg = nbase + nt * 16 + col0;
      float bias = b1[colg];
#pragma unroll
      for (int r = 0; r < 8; ++r) {
        float v = acc[nt][r] + bias;
        v = 0.5f * v * (1.f + erff(v * 0.70710678118654752f));  // exact GELU
        Ys[(mw * 16 + r + hi * 8) * 512 + colg] = f2bf(v);
      }
    }
  }
  __syncthreads();
  int n0 = nw * 64;
  v8f acc2[4] = {zero8(), zero8(), zero8(), zero8()};
  const bf16* arow2 = Ys + (size_t)(mw * 16) * 512;
#pragma unroll
  for (int k0 = 0; k0 < 512; k0 += 32) {
    v16bf a = load_a_frag(arow2 + k0, 512);
    v16bf bbr[4];
#pragma unroll
    for (int nt = 0; nt < 4; ++nt)
      bbr[nt] = load_bt_frag(w2 + (size_t)(n0 + nt * 16) * 512 + k0, 512);
#pragma unroll
    for (int nt = 0; nt < 4; ++nt)
      acc2[nt] = wmma_bf16(a, bbr[nt], acc2[nt]);
  }
#pragma unroll
  for (int nt = 0; nt < 4; ++nt) {
    int colg = n0 + nt * 16 + col0;
    float bias = b2f[colg];
#pragma unroll
    for (int r = 0; r < 8; ++r) {
      size_t rowg = (size_t)m0 + r + hi * 8;
      out[rowg * 128 + colg] = acc2[nt][r] + bias + x1[rowg * 128 + colg];
    }
  }
}

// ---------------------------------------------------------------------------
// launch
// ---------------------------------------------------------------------------
extern "C" void kernel_launch(void* const* d_in, const int* in_sizes, int n_in,
                              void* d_out, int out_size, void* d_ws, size_t ws_size,
                              hipStream_t stream) {
  (void)in_sizes; (void)n_in; (void)out_size; (void)ws_size;
  const float* x        = (const float*)d_in[0];
  const float* maskm    = (const float*)d_in[1];
  const float* n1g      = (const float*)d_in[2];
  const float* n1b      = (const float*)d_in[3];
  const float* qkv_w    = (const float*)d_in[4];
  const float* qkv_b    = (const float*)d_in[5];
  const float* rel_bias = (const float*)d_in[6];
  const float* proj_w   = (const float*)d_in[7];
  const float* proj_b   = (const float*)d_in[8];
  const float* n2g      = (const float*)d_in[9];
  const float* n2b      = (const float*)d_in[10];
  const float* fc1_w    = (const float*)d_in[11];
  const float* fc1_b    = (const float*)d_in[12];
  const float* fc2_w    = (const float*)d_in[13];
  const float* fc2_b    = (const float*)d_in[14];

  char* ws = (char*)d_ws;
  bf16*  wqkv  = (bf16*)(ws + 0);        // 384x128
  bf16*  wproj = (bf16*)(ws + 98304);    // 128x128
  bf16*  w1    = (bf16*)(ws + 131072);   // 512x128
  bf16*  w2    = (bf16*)(ws + 262144);   // 128x512
  float* bqkv  = (float*)(ws + 393216);  // 384 (q-scaled)
  float* btab  = (float*)(ws + 394752);  // 4x98x98
  const size_t XW_OFF  = 1u << 20;
  const size_t XW_SZ   = (size_t)200704 * 128 * 2;   // 51,380,224
  const size_t QKV_OFF = XW_OFF + XW_SZ;
  bf16*  xw     = (bf16*)(ws + XW_OFF);
  bf16*  attout = xw;                                 // Xw dead after QKV GEMM
  bf16*  qkvbuf = (bf16*)(ws + QKV_OFF);              // 200704x384 bf16
  float* x1     = (float*)qkvbuf;                     // QKV dead after attention
  bf16*  h2     = (bf16*)(ws + QKV_OFF + (size_t)200704 * 128 * 4);
  float* out    = (float*)d_out;

  k_prep<<<770, 256, 0, stream>>>(qkv_w, qkv_b, proj_w, fc1_w, fc2_w,
                                  wqkv, bqkv, wproj, w1, w2);
  k_bias<<<38, 256, 0, stream>>>(rel_bias, btab);
  k_ln1<<<25088, 256, 0, stream>>>(x, n1g, n1b, xw);
  k_qkv<<<dim3(3136, 6), 128, 0, stream>>>(xw, wqkv, bqkv, qkvbuf);
  k_attn<<<dim3(2048, 4), 224, 0, stream>>>(qkvbuf, btab, maskm, attout);
  k_proj<<<3136, 256, 0, stream>>>(attout, wproj, proj_b, x, n2g, n2b, x1, h2);
  k_mlp<<<3136, 256, 0, stream>>>(h2, w1, fc1_b, w2, fc2_b, x1, out);
}
